// AttnLSTMEncoder_13469017440708
// MI455X (gfx1250) — matrix operand order repaired and checked by reading, not compiled, not run on hardware
//
#include <hip/hip_runtime.h>
#include <hip/hip_bf16.h>
#include <math.h>

// ---------------------------------------------------------------------------
// AttnLSTMEncoder for MI455X (gfx1250, wave32, WMMA).
//   Phase 1: xg = gather(emb, src) @ Wx^T + bx   (memory-bound big GEMM, bf16 WMMA)
//   Phase 2: persistent cooperative kernel, 512 sequential LSTM steps:
//            gates = xg[:,t,:] + h @ Wh^T (bf16 WMMA, Wh resident in 192MB L2)
//            pointwise LSTM cell update between grid barriers.
// ---------------------------------------------------------------------------

typedef __attribute__((ext_vector_type(16))) __bf16 v16bf;
typedef __attribute__((ext_vector_type(8)))  float  v8f;

#define EMBED   1024
#define HIDDEN  1024
#define BATCH   64
#define SRCLEN  512
#define G4      4096            // 4*HIDDEN
#define MTOT    (BATCH*SRCLEN)  // 32768 rows of the big GEMM
#define REC_BLOCKS 128          // persistent-kernel grid (must be co-resident)

// ---------------- helpers ----------------

__device__ __forceinline__ __bf16 f2bf(float x) {
  // round-to-nearest-even f32 -> bf16
  unsigned u = __builtin_bit_cast(unsigned, x);
  unsigned r = u + 0x7FFFu + ((u >> 16) & 1u);
  unsigned short h = (unsigned short)(r >> 16);
  return __builtin_bit_cast(__bf16, h);
}

__device__ __forceinline__ float sigm(float x) {
  return 1.0f / (1.0f + __expf(-x));
}

__device__ __forceinline__ v8f wmma_bf16(v16bf a, v16bf b, v8f c) {
  // D = A(16x32) * B(32x16) + C(16x16), f32 accumulate
  return __builtin_amdgcn_wmma_f32_16x16x32_bf16(
      /*neg_a=*/false, a, /*neg_b=*/false, b,
      /*c_mod=*/(short)0, c, /*reuse_a=*/false, /*reuse_b=*/false);
}

union AV16 { v16bf v; uint4 u[2]; };

// device-wide two-phase barrier (cooperative persistent kernel)
__device__ __forceinline__ void grid_sync(unsigned* bar, unsigned nblk) {
  __syncthreads();
  if (threadIdx.x == 0) {
    __threadfence();
    unsigned gen = __hip_atomic_load(&bar[1], __ATOMIC_RELAXED, __HIP_MEMORY_SCOPE_AGENT);
    unsigned arrived = __hip_atomic_fetch_add(&bar[0], 1u, __ATOMIC_ACQ_REL, __HIP_MEMORY_SCOPE_AGENT);
    if (arrived == nblk - 1u) {
      __hip_atomic_store(&bar[0], 0u, __ATOMIC_RELAXED, __HIP_MEMORY_SCOPE_AGENT);
      __hip_atomic_fetch_add(&bar[1], 1u, __ATOMIC_RELEASE, __HIP_MEMORY_SCOPE_AGENT);
    } else {
      while (__hip_atomic_load(&bar[1], __ATOMIC_ACQUIRE, __HIP_MEMORY_SCOPE_AGENT) == gen) {
        __builtin_amdgcn_s_sleep(1);
      }
    }
  }
  __syncthreads();
}

// ---------------- kernel 0: per-call init (no cached state allowed) ----------------

__global__ void __launch_bounds__(256)
prep_kernel(const float* __restrict__ Wx, const float* __restrict__ Wh,
            __bf16* __restrict__ Wxbf, __bf16* __restrict__ Whbf,
            float* __restrict__ cbuf, __bf16* __restrict__ hbuf,
            unsigned* __restrict__ bar) {
  const int i = blockIdx.x * blockDim.x + threadIdx.x;
  if (i < G4 * EMBED) {                 // 4M elements each
    Wxbf[i] = f2bf(Wx[i]);
    Whbf[i] = f2bf(Wh[i]);
  }
  if (i < BATCH * HIDDEN) {             // h0 = c0 = 0
    cbuf[i] = 0.0f;
    hbuf[i] = f2bf(0.0f);
  }
  if (i < 2) bar[i] = 0u;               // grid-barrier state
}

// ---------------- kernel 1: xg = gather(emb, src) @ Wx^T + bx ----------------
// Each wave computes a 16(M) x 64(N) tile: 4 accumulators, A reused 4x.
// A rows are embedding gathers (f32 -> bf16 in registers); B reads Wxbf rows
// contiguously (B(k,n) = Wx[n][k]).

__global__ void __launch_bounds__(256)
xg_gemm_kernel(const int* __restrict__ src, const float* __restrict__ emb,
               const __bf16* __restrict__ Wxbf, const float* __restrict__ bx,
               float* __restrict__ xg) {
  const int wave  = (blockIdx.x * blockDim.x + threadIdx.x) >> 5; // 0..131071
  const int lane  = threadIdx.x & 31;
  const int l15   = lane & 15;
  const int lhalf = lane >> 4;
  const int mtile = wave >> 6;    // 0..2047  (M tiles of 16)
  const int n64   = wave & 63;    // 0..63    (N groups of 64)
  const int nbase = n64 * 64;

  // A source: row m = mtile*16 + l15  ->  emb[src[m]]
  const int m     = mtile * 16 + l15;
  const int token = src[m];
  const float* arow = emb + (size_t)token * EMBED + lhalf * 16;

  v8f acc[4] = {v8f{}, v8f{}, v8f{}, v8f{}};

  for (int k = 0; k < EMBED; k += 32) {
    // A tile 16x32 bf16: lane holds chunk [k+16*lhalf, +16)
    AV16 a;
    const float4 f0 = *(const float4*)(arow + k);
    const float4 f1 = *(const float4*)(arow + k + 4);
    const float4 f2 = *(const float4*)(arow + k + 8);
    const float4 f3 = *(const float4*)(arow + k + 12);
    __bf16* ae = (__bf16*)&a;
    ae[0]=f2bf(f0.x); ae[1]=f2bf(f0.y); ae[2]=f2bf(f0.z); ae[3]=f2bf(f0.w);
    ae[4]=f2bf(f1.x); ae[5]=f2bf(f1.y); ae[6]=f2bf(f1.z); ae[7]=f2bf(f1.w);
    ae[8]=f2bf(f2.x); ae[9]=f2bf(f2.y); ae[10]=f2bf(f2.z); ae[11]=f2bf(f2.w);
    ae[12]=f2bf(f3.x); ae[13]=f2bf(f3.y); ae[14]=f2bf(f3.z); ae[15]=f2bf(f3.w);

#pragma unroll
    for (int s = 0; s < 4; ++s) {
      const int nrow = nbase + s * 16 + l15;
      const v16bf b = *(const v16bf*)(Wxbf + (size_t)nrow * EMBED + k + lhalf * 16);
      acc[s] = wmma_bf16(a.v, b, acc[s]);
    }
  }

  // epilogue: C/D layout -> element (v,lane) = (M = v + 8*lhalf, N = l15)
  const int mrowbase = mtile * 16 + lhalf * 8;
#pragma unroll
  for (int s = 0; s < 4; ++s) {
    const int g = nbase + s * 16 + l15;
    const float bias = bx[g];
#pragma unroll
    for (int v = 0; v < 8; ++v) {
      xg[(size_t)(mrowbase + v) * G4 + g] = acc[s][v] + bias;
    }
  }
}

// ---------------- kernel 2: persistent LSTM recurrence ----------------
// 128 blocks x 256 threads = 1024 waves. Each wave owns one 16x16 gate tile:
//   mtile = wave & 3 (batch rows), ntile = wave >> 2 (gate columns).
// Per step: K=1024 bf16 WMMA reduction over h @ Wh^T, add xg[t], store gates,
// grid barrier, pointwise cell update (writes h f32 to out + bf16 for next A),
// grid barrier.

__global__ void __launch_bounds__(256)
lstm_rec_kernel(const float* __restrict__ xg, const __bf16* __restrict__ Whbf,
                float* __restrict__ gates, float* __restrict__ cbuf,
                __bf16* __restrict__ hbuf, float* __restrict__ out,
                unsigned* __restrict__ bar) {
  const int wave  = (blockIdx.x * blockDim.x + threadIdx.x) >> 5; // 0..1023
  const int lane  = threadIdx.x & 31;
  const int l15   = lane & 15;
  const int lhalf = lane >> 4;
  const int mtile = wave & 3;     // 4 tiles cover batch 64
  const int ntile = wave >> 2;    // 256 tiles cover 4096 gate columns

  const __bf16* aptr = hbuf + (size_t)(mtile * 16 + l15) * HIDDEN + lhalf * 16;
  const __bf16* bptr = Whbf + (size_t)(ntile * 16 + l15) * HIDDEN + lhalf * 16;

  const int tid = blockIdx.x * blockDim.x + threadIdx.x;      // 0..32767
  const int nthreads = REC_BLOCKS * 256;

  for (int t = 0; t < SRCLEN; ++t) {
    // ---- gate GEMM: acc = h(16x1024) @ Wh^T slice (1024x16) ----
    v8f acc = v8f{};
#pragma unroll 4
    for (int k = 0; k < HIDDEN; k += 32) {
      AV16 a;
      a.u[0] = *(const uint4*)(aptr + k);        // K = k+16*lhalf .. +8
      a.u[1] = *(const uint4*)(aptr + k + 8);    // K = k+16*lhalf+8 .. +8
      const v16bf b = *(const v16bf*)(bptr + k); // N = l15, K = k+16*lhalf .. +16
      acc = wmma_bf16(a.v, b, acc);
    }

    // ---- add xg[:, t, :] and store pre-activation gates ----
    const int g = ntile * 16 + l15;
    const int bbase = mtile * 16 + lhalf * 8;
#pragma unroll
    for (int v = 0; v < 8; ++v) {
      const int b = bbase + v;   // batch index
      gates[b * G4 + g] = acc[v] + xg[(size_t)(b * SRCLEN + t) * G4 + g];
    }

    grid_sync(bar, REC_BLOCKS);

    // ---- pointwise LSTM cell update: 64*1024 elements over 32768 threads ----
    for (int e = tid; e < BATCH * HIDDEN; e += nthreads) {
      const int b  = e >> 10;
      const int hh = e & 1023;
      const float gi = gates[b * G4 + hh];
      const float gf = gates[b * G4 + HIDDEN + hh];
      const float go = gates[b * G4 + 2 * HIDDEN + hh];
      const float gg = gates[b * G4 + 3 * HIDDEN + hh];
      const float fi = sigm(gi);
      const float ff = sigm(gf);
      const float fo = sigm(go);
      const float fg = tanhf(gg);
      const float c  = ff * cbuf[e] + fi * fg;
      cbuf[e] = c;
      const float h = fo * tanhf(c);
      out[(size_t)(b * SRCLEN + t) * HIDDEN + hh] = h;   // outputs (B,T,H)
      hbuf[e] = f2bf(h);                                 // A operand for t+1
      if (t == SRCLEN - 1) {
        out[(size_t)BATCH * SRCLEN * HIDDEN + e]                    = h; // h_t
        out[(size_t)BATCH * SRCLEN * HIDDEN + BATCH * HIDDEN + e]   = c; // c_t
      }
    }

    grid_sync(bar, REC_BLOCKS);
  }
}

// ---------------- launch ----------------

extern "C" void kernel_launch(void* const* d_in, const int* in_sizes, int n_in,
                              void* d_out, int out_size, void* d_ws, size_t ws_size,
                              hipStream_t stream) {
  const int*   src = (const int*)d_in[0];    // (64,512)
  const float* emb = (const float*)d_in[1];  // (32000,1024)
  const float* Wx  = (const float*)d_in[2];  // (4096,1024)
  const float* bx  = (const float*)d_in[3];  // (4096,)
  const float* Wh  = (const float*)d_in[4];  // (4096,1024)
  float* out = (float*)d_out;                // (B,T,H) ++ h_t ++ c_t

  // workspace layout (all offsets 256B-aligned; total ~530 MB)
  char* ws = (char*)d_ws;
  const size_t XG_BYTES   = (size_t)MTOT * G4 * sizeof(float);     // 512 MB
  const size_t WXB_BYTES  = (size_t)G4 * EMBED * sizeof(__bf16);   // 8 MB
  const size_t WHB_BYTES  = (size_t)G4 * HIDDEN * sizeof(__bf16);  // 8 MB
  const size_t GATE_BYTES = (size_t)BATCH * G4 * sizeof(float);    // 1 MB
  const size_t C_BYTES    = (size_t)BATCH * HIDDEN * sizeof(float);
  const size_t H_BYTES    = (size_t)BATCH * HIDDEN * sizeof(__bf16);

  size_t off = 0;
  float*    xg    = (float*)(ws + off);   off += XG_BYTES;
  __bf16*   Wxbf  = (__bf16*)(ws + off);  off += WXB_BYTES;
  __bf16*   Whbf  = (__bf16*)(ws + off);  off += WHB_BYTES;
  float*    gates = (float*)(ws + off);   off += GATE_BYTES;
  float*    cbuf  = (float*)(ws + off);   off += C_BYTES;
  __bf16*   hbuf  = (__bf16*)(ws + off);  off += H_BYTES;
  unsigned* bar   = (unsigned*)(ws + off);

  // 0) convert weights to bf16, zero h0/c0, reset barrier (every call)
  prep_kernel<<<(G4 * EMBED + 255) / 256, 256, 0, stream>>>(
      Wx, Wh, Wxbf, Whbf, cbuf, hbuf, bar);

  // 1) big GEMM: 131072 waves, each a 16x64 tile of xg (32768 x 4096)
  xg_gemm_kernel<<<(MTOT / 16) * (G4 / 64) * 32 / 256, 256, 0, stream>>>(
      src, emb, Wxbf, bx, xg);

  // 2) persistent recurrence: 128 blocks co-resident, 512 timesteps
  lstm_rec_kernel<<<REC_BLOCKS, 256, 0, stream>>>(
      xg, Whbf, gates, cbuf, hbuf, out, bar);
}
